// Self_Attn_16578573763362
// MI455X (gfx1250) — matrix operand order repaired and checked by reading, not compiled, not run on hardware
//
#include <hip/hip_runtime.h>

typedef __attribute__((ext_vector_type(16))) _Float16 v16h;
typedef __attribute__((ext_vector_type(8)))  _Float16 v8h;
typedef __attribute__((ext_vector_type(8)))  float    v8f;

#define BATCH 4
#define CH    256
#define KIN   32
#define NPIX  4096
#define SROW  4100   // 4096 + 4 floats pad (kills 16-way bank conflicts on column reads)

// --------------------------------------------------------------------------
// Kernel 1: 1x1-conv projections f,g,h  (f32 in -> f16 out, fp32 accumulate)
// grid (NPIX/128, BATCH), 256 threads, dynamic LDS = 256*128*4 B (x tile)
// f,g layout: [b][n][k] (k contiguous -> WMMA A/B fragment loads are 16B/32B)
// h   layout: [b][k][n] (n contiguous -> B fragment of attn@V GEMM)
// --------------------------------------------------------------------------
__global__ __launch_bounds__(256)
void proj_kernel(const float* __restrict__ x,
                 const float* __restrict__ Wf, const float* __restrict__ bf,
                 const float* __restrict__ Wg, const float* __restrict__ bg,
                 const float* __restrict__ Wh, const float* __restrict__ bh,
                 _Float16* __restrict__ fH, _Float16* __restrict__ gH,
                 _Float16* __restrict__ hH)
{
    extern __shared__ float xs[];             // [256 c][128 p]
    const int t  = threadIdx.x;
    const int b  = blockIdx.y;
    const int n0 = blockIdx.x * 128;

    const float* xb = x + (size_t)b * CH * NPIX + n0;
    for (int lin = t; lin < CH * 128; lin += 256) {
        int c = lin >> 7, p = lin & 127;
        xs[c * 128 + p] = xb[(size_t)c * NPIX + p];
    }
    __syncthreads();

    const int p  = t & 127;
    const int kh = t >> 7;        // 0..1 (uniform per wave)
    const int k0 = kh * 16;

    const float* Ws[3] = {Wf, Wg, Wh};
    const float* bs[3] = {bf, bg, bh};
    float acc[16];

    for (int pr = 0; pr < 3; ++pr) {
        const float* W = Ws[pr];
        #pragma unroll
        for (int kk = 0; kk < 16; ++kk) acc[kk] = 0.f;
        for (int c = 0; c < CH; ++c) {
            float xv = xs[c * 128 + p];
            #pragma unroll
            for (int kk = 0; kk < 16; ++kk)
                acc[kk] = __builtin_fmaf(W[(k0 + kk) * CH + c], xv, acc[kk]);
        }
        if (pr < 2) {
            _Float16* out = (pr == 0 ? fH : gH)
                          + ((size_t)b * NPIX + n0 + p) * KIN + k0;
            #pragma unroll
            for (int kk = 0; kk < 16; ++kk)
                out[kk] = (_Float16)(acc[kk] + bs[pr][k0 + kk]);
        } else {
            #pragma unroll
            for (int kk = 0; kk < 16; ++kk)
                hH[((size_t)b * KIN + k0 + kk) * NPIX + n0 + p] =
                    (_Float16)(acc[kk] + bh[k0 + kk]);
        }
    }
}

// --------------------------------------------------------------------------
// Kernel 2: fused attention for a 16-row query tile.
//   s = f_tile^T @ g        (WMMA f16 16x16x32, f32 acc; s kept in 256KB LDS)
//   softmax rows in LDS; write attn^T (attn_map output, coalesced)
//   O = attn @ h^T          (WMMA, 8 waves split K=4096, ds_add_f32 combine)
//   y = gamma * (Wv @ O^T + bv) + x
// grid (NPIX/16, BATCH), 256 threads, dynamic LDS ~= 299 KB
// --------------------------------------------------------------------------
__global__ __launch_bounds__(256)
void attn_kernel(const float* __restrict__ x,
                 const _Float16* __restrict__ fH,
                 const _Float16* __restrict__ gH,
                 const _Float16* __restrict__ hH,
                 const float* __restrict__ Wv, const float* __restrict__ bv,
                 const float* __restrict__ gamma,
                 float* __restrict__ y, float* __restrict__ attn_out)
{
    extern __shared__ float smem[];
    float* sS  = smem;                 // 16 * SROW   : score/attn tile
    float* red = sS  + 16 * SROW;      // 16 * 17     : softmax reductions
    float* Ol  = red + 16 * 17;        // 16 * 33     : O tile (padded)
    float* WvS = Ol  + 16 * 33;        // 256 * 32    : Wv cache
    float* bvS = WvS + CH * KIN;       // 256

    const int t    = threadIdx.x;
    const int b    = blockIdx.y;
    const int n0   = blockIdx.x * 16;
    const int wave = t >> 5;
    const int lane = t & 31;
    const int lm   = lane & 15;        // column / M-row lane index
    const int hs   = lane >> 4;        // lane half select

    for (int i = t; i < CH * KIN; i += 256) WvS[i] = Wv[i];
    for (int i = t; i < CH;       i += 256) bvS[i] = bv[i];
    for (int i = t; i < 16 * 33;  i += 256) Ol[i]  = 0.f;

    // ---------------- GEMM1: s[16 x 4096] = f_tile^T @ g -----------------
    const _Float16* fB = fH + (size_t)b * NPIX * KIN;
    const _Float16* gB = gH + (size_t)b * NPIX * KIN;

    // A fragment (16x32 f16): lane half hs holds K = {8hs..8hs+7, 16+8hs..}
    v8h alo = *(const v8h*)(fB + (size_t)(n0 + lm) * KIN + 8 * hs);
    v8h ahi = *(const v8h*)(fB + (size_t)(n0 + lm) * KIN + 16 + 8 * hs);
    v16h a = __builtin_shufflevector(alo, ahi,
                                     0,1,2,3,4,5,6,7,8,9,10,11,12,13,14,15);

    for (int mt = 0; mt < 32; ++mt) {            // each wave: 32 col-tiles
        int m0 = (wave * 32 + mt) * 16;
        // B fragment (32x16): lane = col, K = {8hs..8hs+7, 16+8hs..}
        v8h blo = *(const v8h*)(gB + (size_t)(m0 + lm) * KIN + 8 * hs);
        v8h bhi = *(const v8h*)(gB + (size_t)(m0 + lm) * KIN + 16 + 8 * hs);
        v16h bm = __builtin_shufflevector(blo, bhi,
                                          0,1,2,3,4,5,6,7,8,9,10,11,12,13,14,15);
        v8f acc = {};
        acc = __builtin_amdgcn_wmma_f32_16x16x32_f16(
                  false, a, false, bm, (short)0, acc, false, false);
        #pragma unroll
        for (int v = 0; v < 8; ++v)
            sS[(v + 8 * hs) * SROW + m0 + lm] = acc[v];
    }
    __syncthreads();

    // ---------------- softmax over each of the 16 rows --------------------
    const int r = t & 15;              // row handled by this thread
    const int g = t >> 4;              // 16 threads per row
    float* row = sS + r * SROW;

    float lmax = -3.0e38f;
    for (int j = 0; j < 256; ++j) lmax = fmaxf(lmax, row[g + 16 * j]);
    red[r * 17 + g] = lmax;
    __syncthreads();
    float rmax = -3.0e38f;
    #pragma unroll
    for (int i = 0; i < 16; ++i) rmax = fmaxf(rmax, red[r * 17 + i]);
    __syncthreads();

    float lsum = 0.f;
    for (int j = 0; j < 256; ++j) {
        int idx = g + 16 * j;
        float e = __expf(row[idx] - rmax);
        row[idx] = e;
        lsum += e;
    }
    red[r * 17 + g] = lsum;
    __syncthreads();
    float rsum = 0.f;
    #pragma unroll
    for (int i = 0; i < 16; ++i) rsum += red[r * 17 + i];
    const float inv = 1.0f / rsum;

    // normalize in LDS + write attn_map[b][m][n] = attn[n][m] (coalesced in r)
    float* aout = attn_out + (size_t)b * NPIX * NPIX + n0 + r;
    for (int j = 0; j < 256; ++j) {
        int m = g + 16 * j;
        float v = row[m] * inv;
        row[m] = v;
        aout[(size_t)m * NPIX] = v;
    }
    __syncthreads();

    // ---------------- GEMM2: O[16 x 32] = attn @ h^T ----------------------
    const _Float16* hB = hH + (size_t)b * KIN * NPIX;
    v8f o0 = {}, o1 = {};
    const int cstart = wave * 512;               // 8 waves split K = 4096
    for (int it = 0; it < 16; ++it) {
        int cb = cstart + it * 32;
        v16h aa;                                  // A from normalized attn in LDS
        #pragma unroll
        for (int e = 0; e < 8; ++e)
            aa[e] = (_Float16)sS[lm * SROW + cb + 16 * hs + e];
        #pragma unroll
        for (int e = 8; e < 16; ++e)
            aa[e] = (_Float16)sS[lm * SROW + cb + 16 * hs + e];
        v16h b0 = *(const v16h*)(hB + (size_t)(lm)      * NPIX + cb + 16 * hs);
        v16h b1 = *(const v16h*)(hB + (size_t)(16 + lm) * NPIX + cb + 16 * hs);
        o0 = __builtin_amdgcn_wmma_f32_16x16x32_f16(
                 false, aa, false, b0, (short)0, o0, false, false);
        o1 = __builtin_amdgcn_wmma_f32_16x16x32_f16(
                 false, aa, false, b1, (short)0, o1, false, false);
    }
    #pragma unroll
    for (int v = 0; v < 8; ++v) {
        atomicAdd(&Ol[(v + 8 * hs) * 33 + lm],      o0[v]);   // ds_add_f32
        atomicAdd(&Ol[(v + 8 * hs) * 33 + 16 + lm], o1[v]);
    }
    __syncthreads();

    // ---------------- output projection + residual ------------------------
    const float gm = gamma[0];
    const float* xb = x + (size_t)b * CH * NPIX;
    float* yb = y + (size_t)b * CH * NPIX;
    #pragma unroll
    for (int j = 0; j < 16; ++j) {
        int co = g + 16 * j;
        float o = bvS[co];
        #pragma unroll
        for (int k = 0; k < KIN; ++k)
            o = __builtin_fmaf(WvS[co * KIN + k], Ol[r * 33 + k], o);
        size_t idx = (size_t)co * NPIX + n0 + r;   // consecutive r -> coalesced
        yb[idx] = gm * o + xb[idx];
    }
}

// --------------------------------------------------------------------------
extern "C" void kernel_launch(void* const* d_in, const int* in_sizes, int n_in,
                              void* d_out, int out_size, void* d_ws, size_t ws_size,
                              hipStream_t stream)
{
    const float* x     = (const float*)d_in[0];
    const float* Wf    = (const float*)d_in[1];
    const float* bf    = (const float*)d_in[2];
    const float* Wg    = (const float*)d_in[3];
    const float* bg    = (const float*)d_in[4];
    const float* Wh    = (const float*)d_in[5];
    const float* bh    = (const float*)d_in[6];
    const float* Wv    = (const float*)d_in[7];
    const float* bv    = (const float*)d_in[8];
    const float* gamma = (const float*)d_in[9];

    _Float16* fH = (_Float16*)d_ws;                       // 1 MB each
    _Float16* gH = fH + (size_t)BATCH * NPIX * KIN;
    _Float16* hH = gH + (size_t)BATCH * NPIX * KIN;

    float* y    = (float*)d_out;
    float* attn = y + (size_t)BATCH * CH * NPIX;

    dim3 g1(NPIX / 128, BATCH);
    size_t sh1 = (size_t)CH * 128 * sizeof(float);        // 128 KB
    proj_kernel<<<g1, 256, sh1, stream>>>(x, Wf, bf, Wg, bg, Wh, bh, fH, gH, hH);

    dim3 g2(NPIX / 16, BATCH);
    size_t sh2 = (size_t)(16 * SROW + 16 * 17 + 16 * 33 + CH * KIN + CH)
               * sizeof(float);                           // ~299 KB of the 320 KB LDS
    attn_kernel<<<g2, 256, sh2, stream>>>(x, fH, gH, hH, Wv, bv, gamma, y, attn);
}
